// LGCR_ver2_66374424592495
// MI455X (gfx1250) — compile-verified, matrix operand order checked by reading, not run on hardware
//
#include <hip/hip_runtime.h>

typedef __attribute__((ext_vector_type(16))) _Float16 v16h;
typedef __attribute__((ext_vector_type(8)))  _Float16 v8h;
typedef __attribute__((ext_vector_type(8)))  float    v8f;

#define B_  32
#define C_  256
#define H_  56
#define W_  56
#define R_  64
#define HW_ 3136
#define NT_ 196   // HW_/16 column tiles

// ---------------------------------------------------------------------------
// Workspace layout (bytes):
//   gapC  : 0        .. 32768   (B*C   f32 plane sums)
//   gapH  : 32768    .. 39936   (B*H   f32 accum)
//   gapW  : 39936    .. 47104   (B*W   f32 accum)
//   cOutH : 47104    .. 1095680 (B*C*R f16, row = 64 halfs = 128B aligned)
//   hOutH : 1095680  .. 1325056 (B*R*H f16)
//   wOutH : 1325056  .. 1554432 (B*R*W f16)
// ---------------------------------------------------------------------------

__global__ void zero_kernel(float* p, int n) {
    int i = blockIdx.x * blockDim.x + threadIdx.x;
    if (i < n) p[i] = 0.f;
}

// One block per (b,c) plane: plane sum -> gapC, row/col sums -> gapH/gapW.
__global__ __launch_bounds__(256)
void reduce_kernel(const float* __restrict__ x,
                   float* __restrict__ gapC,
                   float* __restrict__ gapH,
                   float* __restrict__ gapW) {
    __shared__ float sH[H_];
    __shared__ float sW[W_];
    int bc  = blockIdx.x;
    int b   = bc / C_;
    int tid = threadIdx.x;
    if (tid < H_) sH[tid] = 0.f;
    if (tid < W_) sW[tid] = 0.f;
    __syncthreads();
    const float* plane = x + (size_t)bc * HW_;
    for (int i = tid; i < HW_; i += 256) {
        float v = plane[i];
        int h = i / W_, w = i - h * W_;
        atomicAdd(&sH[h], v);
        atomicAdd(&sW[w], v);
    }
    __syncthreads();
    if (tid < H_) atomicAdd(&gapH[b * H_ + tid], sH[tid]);
    if (tid < W_) atomicAdd(&gapW[b * W_ + tid], sW[tid]);
    if (tid == 0) {
        float s = 0.f;
        for (int h = 0; h < H_; ++h) s += sH[h];
        gapC[bc] = s;
    }
}

// conv1d(k=3, SAME) + PReLU for the three factor branches; outputs f16.
__global__ void branch_kernel(const float* __restrict__ gapC,
                              const float* __restrict__ gapH,
                              const float* __restrict__ gapW,
                              const float* __restrict__ Wc, const float* __restrict__ ac_p,
                              const float* __restrict__ Ww, const float* __restrict__ aw_p,
                              const float* __restrict__ Wh, const float* __restrict__ ah_p,
                              _Float16* __restrict__ cOutH,
                              _Float16* __restrict__ hOutH,
                              _Float16* __restrict__ wOutH) {
    int idx = blockIdx.x * blockDim.x + threadIdx.x;
    const int N1 = B_ * C_ * R_;
    const int N2 = B_ * R_ * H_;
    const int N3 = B_ * R_ * W_;
    if (idx < N1) {
        int b = idx / (C_ * R_);
        int rem = idx - b * (C_ * R_);
        int c = rem / R_, r = rem - (rem / R_) * R_;
        const float inv = 1.f / (float)HW_;
        float x0 = (c - 1 >= 0) ? gapC[b * C_ + c - 1] * inv : 0.f;
        float x1 = gapC[b * C_ + c] * inv;
        float x2 = (c + 1 < C_) ? gapC[b * C_ + c + 1] * inv : 0.f;
        float acc = Wc[r * 3] * x0 + Wc[r * 3 + 1] * x1 + Wc[r * 3 + 2] * x2;
        float a = *ac_p;
        cOutH[(size_t)(b * C_ + c) * R_ + r] = (_Float16)(acc >= 0.f ? acc : a * acc);
    } else if (idx < N1 + N2) {
        int i = idx - N1;
        int b = i / (R_ * H_);
        int rem = i - b * (R_ * H_);
        int r = rem / H_, h = rem - (rem / H_) * H_;
        const float inv = 1.f / (float)(C_ * W_);
        float x0 = (h - 1 >= 0) ? gapH[b * H_ + h - 1] * inv : 0.f;
        float x1 = gapH[b * H_ + h] * inv;
        float x2 = (h + 1 < H_) ? gapH[b * H_ + h + 1] * inv : 0.f;
        float acc = Wh[r * 3] * x0 + Wh[r * 3 + 1] * x1 + Wh[r * 3 + 2] * x2;
        float a = *ah_p;
        hOutH[(size_t)(b * R_ + r) * H_ + h] = (_Float16)(acc >= 0.f ? acc : a * acc);
    } else if (idx < N1 + N2 + N3) {
        int i = idx - N1 - N2;
        int b = i / (R_ * W_);
        int rem = i - b * (R_ * W_);
        int r = rem / W_, w = rem - (rem / W_) * W_;
        const float inv = 1.f / (float)(C_ * H_);
        float x0 = (w - 1 >= 0) ? gapW[b * W_ + w - 1] * inv : 0.f;
        float x1 = gapW[b * W_ + w] * inv;
        float x2 = (w + 1 < W_) ? gapW[b * W_ + w + 1] * inv : 0.f;
        float acc = Ww[r * 3] * x0 + Ww[r * 3 + 1] * x1 + Ww[r * 3 + 2] * x2;
        float a = *aw_p;
        wOutH[(size_t)(b * R_ + r) * W_ + w] = (_Float16)(acc >= 0.f ? acc : a * acc);
    }
}

// Build a v16h WMMA 16-bit fragment from two contiguous 8-half chunks.
// ISA 16-bit A/B layout: lane (grp = 0|8) holds K = grp..grp+7 in VGPRs 0-3
// and K = 16+grp..16+grp+7 in VGPRs 4-7 -> exactly chunks [base, base+8) and
// [base+16, base+24) of a contiguous K-row.
__device__ __forceinline__ v16h frag16(v8h lo, v8h hi) {
    v16h f;
#pragma unroll
    for (int j = 0; j < 8; ++j) { f[j] = lo[j]; f[8 + j] = hi[j]; }
    return f;
}

// Fused: WMMA GEMM (c_out x [h⊗w]) -> channel softmax -> 3x3x3 conv gate + residual.
// Grid: B_*NT_ workgroups of 256 threads (8 wave32s). Each WG owns one batch
// and 16 flattened hw-columns, for all 256 channels.
__global__ __launch_bounds__(256)
void main_kernel(const float* __restrict__ x,
                 const _Float16* __restrict__ cOutH,
                 const _Float16* __restrict__ hOutH,
                 const _Float16* __restrict__ wOutH,
                 const float* __restrict__ W3d,
                 float* __restrict__ out) {
    __shared__ _Float16 ldsB[16 * R_];   // B tile TRANSPOSED: [n][k], 128B rows
    __shared__ float ldsN[C_ * 17];      // nchw strip, padded stride (17)
    __shared__ float ldsRed[256];
    __shared__ float colMax[16];
    __shared__ float colInv[16];
    __shared__ float w3s[27];

    int tid  = threadIdx.x;
    int blk  = blockIdx.x;
    int b    = blk / NT_;
    int tile = blk - b * NT_;
    int n0   = tile * 16;

    if (tid < 27) w3s[tid] = W3d[tid];

    // ---- Phase 1: build B tile  ldsB[n][k] = h_out[b,k,h(n)] * w_out[b,k,w(n)]
    // Packed 32-bit stores (two halfs per thread-item).
    for (int i = tid; i < 512; i += 256) {
        int n  = i >> 5;            // 0..15
        int kk = i & 31;            // pair index
        int k2 = kk * 2;
        int pos = n0 + n;
        int hh = pos / W_, ww = pos - hh * W_;
        float h0 = (float)hOutH[(size_t)(b * R_ + k2) * H_ + hh];
        float w0 = (float)wOutH[(size_t)(b * R_ + k2) * W_ + ww];
        float h1 = (float)hOutH[(size_t)(b * R_ + k2 + 1) * H_ + hh];
        float w1 = (float)wOutH[(size_t)(b * R_ + k2 + 1) * W_ + ww];
        union { _Float16 h[2]; unsigned u; } pk;
        pk.h[0] = (_Float16)(h0 * w0);
        pk.h[1] = (_Float16)(h1 * w1);
        ((unsigned*)ldsB)[n * 32 + kk] = pk.u;
    }
    __syncthreads();

    // ---- Phase 2: WMMA. 8 waves x 2 M-tiles each = 256 channels.
    int lane = tid & 31;
    int wave = tid >> 5;
    int grp  = (lane >= 16) ? 8 : 0;   // K offset for upper half-wave (ISA layout)
    int ncol = lane & 15;

    // B fragments: 4 x ds_load_b128 per lane (16B-aligned chunks of row ncol).
    const v8h* bp = (const v8h*)(ldsB + ncol * R_ + grp);
    v8h bch0 = bp[0];   // k: grp   .. grp+7
    v8h bch1 = bp[2];   // k: 16+grp.. 16+grp+7
    v8h bch2 = bp[4];   // k: 32+grp.. 32+grp+7
    v8h bch3 = bp[6];   // k: 48+grp.. 48+grp+7
    v16h bm0 = frag16(bch0, bch1);
    v16h bm1 = frag16(bch2, bch3);

#pragma unroll
    for (int s = 0; s < 2; ++s) {
        int mt = wave * 2 + s;
        int mrow = mt * 16 + ncol;   // A: lanes 0-15 and 16-31 both hold M=0..15
        const _Float16* crow = cOutH + (size_t)(b * C_ + mrow) * R_;
        // A fragments: 4 x global_load_b128 per lane.
        const v8h* ap = (const v8h*)(crow + grp);
        v8h ach0 = ap[0], ach1 = ap[2], ach2 = ap[4], ach3 = ap[6];
        v16h am0 = frag16(ach0, ach1);
        v16h am1 = frag16(ach2, ach3);

        v8f acc = {};
        acc = __builtin_amdgcn_wmma_f32_16x16x32_f16(false, am0, false, bm0,
                                                     (short)0, acc, false, false);
        acc = __builtin_amdgcn_wmma_f32_16x16x32_f16(false, am1, false, bm1,
                                                     (short)0, acc, false, false);
#pragma unroll
        for (int v = 0; v < 8; ++v) {
            int ch = mt * 16 + v + ((lane >= 16) ? 8 : 0);  // C/D layout
            ldsN[ch * 17 + ncol] = acc[v];
        }
    }
    __syncthreads();

    // ---- Phase 3: softmax stats over 256 channels per column.
    int n = tid & 15, seg = tid >> 4;
    float mx = -3.4e38f;
#pragma unroll
    for (int i = 0; i < 16; ++i) mx = fmaxf(mx, ldsN[(seg * 16 + i) * 17 + n]);
    ldsRed[seg * 16 + n] = mx;
    __syncthreads();
    if (tid < 16) {
        float m2 = -3.4e38f;
        for (int s2 = 0; s2 < 16; ++s2) m2 = fmaxf(m2, ldsRed[s2 * 16 + tid]);
        colMax[tid] = m2;
    }
    __syncthreads();
    float cm = colMax[n];
    float sm = 0.f;
#pragma unroll
    for (int i = 0; i < 16; ++i) sm += __expf(ldsN[(seg * 16 + i) * 17 + n] - cm);
    ldsRed[seg * 16 + n] = sm;
    __syncthreads();
    if (tid < 16) {
        float sv = 0.f;
        for (int s2 = 0; s2 < 16; ++s2) sv += ldsRed[s2 * 16 + tid];
        colInv[tid] = 1.f / sv;
    }
    __syncthreads();

    // ---- Phase 4: sliding 3x3x3 conv over channels + gated residual output.
    // Thread t: column n = t&15, channels [seg*16, seg*16+16).
    int pos = n0 + n;
    int hh = pos / W_, ww = pos - hh * W_;
    int cbase = seg * 16;
    float cmv = colMax[n], civ = colInv[n];
    const size_t bOff = (size_t)b * C_ * HW_;

    float r_m1 = 0.f, r_0 = 0.f, r_p1 = 0.f, cprev = 0.f;
    for (int cc = cbase - 1; cc <= cbase + 16; ++cc) {
        float sp[9];
        bool cok = (cc >= 0) && (cc < C_);
#pragma unroll
        for (int dh = 0; dh < 3; ++dh) {
            int h2 = hh + dh - 1;
#pragma unroll
            for (int dw = 0; dw < 3; ++dw) {
                int w2 = ww + dw - 1;
                bool ok = cok && (h2 >= 0) && (h2 < H_) && (w2 >= 0) && (w2 < W_);
                sp[dh * 3 + dw] = ok ? x[bOff + (size_t)cc * HW_ + h2 * W_ + w2] : 0.f;
            }
        }
        float d0 = 0.f, d1 = 0.f, d2 = 0.f;
#pragma unroll
        for (int i = 0; i < 9; ++i) {
            d0 += w3s[i]      * sp[i];
            d1 += w3s[9 + i]  * sp[i];
            d2 += w3s[18 + i] * sp[i];
        }
        r_m1 += d2; r_0 += d1; r_p1 += d0;
        int e = cc - 1;
        if (e >= cbase && e < cbase + 16) {
            float g = __expf(ldsN[e * 17 + n] - cmv) * civ;
            out[bOff + (size_t)e * HW_ + pos] = r_m1 * g + cprev;
        }
        cprev = sp[4];
        r_m1 = r_0; r_0 = r_p1; r_p1 = 0.f;
    }
}

extern "C" void kernel_launch(void* const* d_in, const int* in_sizes, int n_in,
                              void* d_out, int out_size, void* d_ws, size_t ws_size,
                              hipStream_t stream) {
    (void)in_sizes; (void)n_in; (void)out_size; (void)ws_size;
    const float* x   = (const float*)d_in[0];
    const float* Wc  = (const float*)d_in[1];
    const float* ac  = (const float*)d_in[2];
    const float* Ww  = (const float*)d_in[3];
    const float* aw  = (const float*)d_in[4];
    const float* Wh  = (const float*)d_in[5];
    const float* ah  = (const float*)d_in[6];
    const float* W3d = (const float*)d_in[7];
    float* out = (float*)d_out;

    char* ws = (char*)d_ws;
    float*    gapC  = (float*)(ws + 0);
    float*    gapH  = (float*)(ws + 32768);
    float*    gapW  = (float*)(ws + 39936);
    _Float16* cOutH = (_Float16*)(ws + 47104);
    _Float16* hOutH = (_Float16*)(ws + 1095680);
    _Float16* wOutH = (_Float16*)(ws + 1325056);

    // gapH and gapW are contiguous: zero both (2 * 1792 floats).
    zero_kernel<<<(3584 + 255) / 256, 256, 0, stream>>>(gapH, 3584);
    reduce_kernel<<<B_ * C_, 256, 0, stream>>>(x, gapC, gapH, gapW);
    int Ntot = B_ * C_ * R_ + 2 * B_ * R_ * 56;
    branch_kernel<<<(Ntot + 255) / 256, 256, 0, stream>>>(
        gapC, gapH, gapW, Wc, ac, Ww, aw, Wh, ah, cOutH, hOutH, wOutH);
    main_kernel<<<B_ * NT_, 256, 0, stream>>>(x, cOutH, hOutH, wOutH, W3d, out);
}
